// VelocityEncoder_54039278518831
// MI455X (gfx1250) — matrix operand order repaired and checked by reading, not compiled, not run on hardware
//
#include <hip/hip_runtime.h>
#include <hip/hip_bf16.h>

typedef __attribute__((ext_vector_type(16))) __bf16 v16bf;
typedef __attribute__((ext_vector_type(8)))  float  v8f;

#define B_DIM 64
#define A_DIM 1024
#define E_DIM 256

// ---------------------------------------------------------------------------
// Kernel 0: split W2 (256x256 f32, row e, col f) into bf16 hi/lo, prepacked in
// WMMA B fragments for 16x16x32 bf16 (this file's contiguous k order):
//   frag index = (ntile*8 + ktile), within frag: lane holds column N = 16*ntile
//   + (lane&15); elems 0..15 = K = 32*ktile + 16*(lane>>4) + elem.
// Each (frag,lane) = 16 consecutive bf16 = 32 bytes -> two b128 loads later.
// ---------------------------------------------------------------------------
__global__ __launch_bounds__(256) void prepack_w2_kernel(
    const float* __restrict__ W2, unsigned short* __restrict__ hi,
    unsigned short* __restrict__ lo) {
  int t = blockIdx.x * 256 + threadIdx.x;          // 0 .. 65535
  int elem = t & 15;
  int lane = (t >> 4) & 31;
  int kt   = (t >> 9) & 7;
  int nt   = t >> 12;
  int K = kt * 32 + ((lane >> 4) << 4) + elem;
  int N = nt * 16 + (lane & 15);
  float v = W2[K * E_DIM + N];
  __bf16 h = (__bf16)v;
  float  r = v - (float)h;
  __bf16 l = (__bf16)r;
  hi[t] = __builtin_bit_cast(unsigned short, h);
  lo[t] = __builtin_bit_cast(unsigned short, l);
}

// insert (x, xi) into the sorted-ascending 4-list (v0..v3 / i0..i3)
#define TOP4_INSERT(x, xi)                                                  \
  do {                                                                      \
    if ((x) < v3) {                                                         \
      v3 = (x); i3 = (xi);                                                  \
      if (v3 < v2) { float tf = v2; v2 = v3; v3 = tf; int ti = i2; i2 = i3; i3 = ti; } \
      if (v2 < v1) { float tf = v1; v1 = v2; v2 = tf; int ti = i1; i1 = i2; i2 = ti; } \
      if (v1 < v0) { float tf = v0; v0 = v1; v1 = tf; int ti = i0; i0 = i1; i1 = ti; } \
    }                                                                       \
  } while (0)

// ---------------------------------------------------------------------------
// Main fused kernel: 1 workgroup = 16 waves = one 16-row M-tile.
// Wave w: top-4 / features / GEMM1 / LayerNorm for row (tile*16 + w),
// then owns N-tile w of GEMM2 via bf16x3 WMMA.
// ---------------------------------------------------------------------------
__global__ __launch_bounds__(512) void velocity_encoder_kernel(
    const float* __restrict__ vel,          // (B, A, 3)
    const float* __restrict__ dist,         // (B, A, A)
    const float* __restrict__ W1,           // (6, E)
    const float* __restrict__ b1,           // (E)
    const float* __restrict__ gamma,        // (E)
    const float* __restrict__ beta,         // (E)
    const float* __restrict__ b2,           // (E)
    const unsigned short* __restrict__ w2hi,// prepacked B-frags
    const unsigned short* __restrict__ w2lo,
    float* __restrict__ out) {              // (B, A, E)
  const int tid  = threadIdx.x;
  const int lane = tid & 31;
  const int w    = tid >> 5;                // wave id 0..15
  const int tile = blockIdx.x;              // 0..4095
  const int row  = tile * 16 + w;           // global atom row
  const int b    = row >> 10;
  const int a    = row & (A_DIM - 1);

  __shared__ alignas(16) __bf16 sHi[16 * E_DIM];
  __shared__ alignas(16) __bf16 sLo[16 * E_DIM];

  // ---- top-4 smallest over the 1024-wide distance row (streaming, NT) ----
  const float* drow = dist + (size_t)row * A_DIM;
  float v0 = 3.4e38f, v1 = 3.4e38f, v2 = 3.4e38f, v3 = 3.4e38f;
  int   i0 = 0, i1 = 0, i2 = 0, i3 = 0;
#pragma unroll 4
  for (int it = 0; it < 32; ++it) {
    int idx = lane + (it << 5);
    float d = __builtin_nontemporal_load(drow + idx);
    TOP4_INSERT(d, idx);
  }
  // butterfly merge: after 5 rounds every lane holds the global top-4
#pragma unroll
  for (int off = 1; off < 32; off <<= 1) {
    float w0 = __shfl_xor(v0, off), w1 = __shfl_xor(v1, off);
    float w2v = __shfl_xor(v2, off), w3 = __shfl_xor(v3, off);
    int j0 = __shfl_xor(i0, off), j1 = __shfl_xor(i1, off);
    int j2 = __shfl_xor(i2, off), j3 = __shfl_xor(i3, off);
    TOP4_INSERT(w0, j0);
    TOP4_INSERT(w1, j1);
    TOP4_INSERT(w2v, j2);
    TOP4_INSERT(w3, j3);
  }
  // neighbors = 2nd..4th smallest (drop overall smallest, as idx[...,1:])

  // ---- features: [v, v - mean(v_nbr)] ----
  const float* vb = vel + (size_t)b * (A_DIM * 3);
  float cx = vb[a * 3 + 0], cy = vb[a * 3 + 1], cz = vb[a * 3 + 2];
  const float third = 1.0f / 3.0f;
  float nx = (vb[i1 * 3 + 0] + vb[i2 * 3 + 0] + vb[i3 * 3 + 0]) * third;
  float ny = (vb[i1 * 3 + 1] + vb[i2 * 3 + 1] + vb[i3 * 3 + 1]) * third;
  float nz = (vb[i1 * 3 + 2] + vb[i2 * 3 + 2] + vb[i3 * 3 + 2]) * third;
  float c[6] = {cx, cy, cz, cx - nx, cy - ny, cz - nz};

  // ---- GEMM1 (6 -> 256) + ReLU, 8 outputs per lane ----
  float h[8];
  float s = 0.f, q = 0.f;
#pragma unroll
  for (int j = 0; j < 8; ++j) {
    int e = lane + (j << 5);
    float acc = b1[e];
#pragma unroll
    for (int f = 0; f < 6; ++f) acc = fmaf(c[f], W1[f * E_DIM + e], acc);
    acc = fmaxf(acc, 0.f);
    h[j] = acc;
    s += acc;
    q += acc * acc;
  }
  // ---- LayerNorm over E=256 (wave reduction) ----
#pragma unroll
  for (int off = 16; off; off >>= 1) {
    s += __shfl_xor(s, off);
    q += __shfl_xor(q, off);
  }
  float mean = s * (1.0f / 256.0f);
  float var  = q * (1.0f / 256.0f) - mean * mean;
  float rinv = __frsqrt_rn(var + 1e-5f);
#pragma unroll
  for (int j = 0; j < 8; ++j) {
    int e = lane + (j << 5);
    float g = (h[j] - mean) * rinv * gamma[e] + beta[e];
    __bf16 gh = (__bf16)g;
    sHi[w * E_DIM + e] = gh;
    sLo[w * E_DIM + e] = (__bf16)(g - (float)gh);
  }
  __syncthreads();

  // ---- GEMM2: wave w owns N-tile w; bf16x3 split-precision WMMA ----
  v8f acc = {0.f, 0.f, 0.f, 0.f, 0.f, 0.f, 0.f, 0.f};
  const int nt  = w;
  const int M   = lane & 15;
  const int klo = (lane & 16) ? 16 : 0;  // k order used here: lane-half K contiguous, same as B prepack
#pragma unroll
  for (int kt = 0; kt < 8; ++kt) {
    union { v16bf v; unsigned int u[8]; } ah, al;
    int base = M * E_DIM + kt * 32 + klo;
#pragma unroll
    for (int p = 0; p < 4; ++p) {
      ah.u[p]     = *(const unsigned int*)&sHi[base + 2 * p];
      ah.u[4 + p] = *(const unsigned int*)&sHi[base + 8 + 2 * p];
      al.u[p]     = *(const unsigned int*)&sLo[base + 2 * p];
      al.u[4 + p] = *(const unsigned int*)&sLo[base + 8 + 2 * p];
    }
    union { v16bf v; uint4 q[2]; } bh, bl;
    const uint4* ph = (const uint4*)w2hi + ((size_t)(nt * 8 + kt) * 32 + lane) * 2;
    const uint4* pl = (const uint4*)w2lo + ((size_t)(nt * 8 + kt) * 32 + lane) * 2;
    bh.q[0] = ph[0]; bh.q[1] = ph[1];
    bl.q[0] = pl[0]; bl.q[1] = pl[1];
    // hi*hi + hi*lo + lo*hi  (drop lo*lo; ~fp32 accuracy)
    acc = __builtin_amdgcn_wmma_f32_16x16x32_bf16(false, ah.v, false, bh.v,
                                                  (short)0, acc, false, false);
    acc = __builtin_amdgcn_wmma_f32_16x16x32_bf16(false, ah.v, false, bl.v,
                                                  (short)0, acc, false, false);
    acc = __builtin_amdgcn_wmma_f32_16x16x32_bf16(false, al.v, false, bh.v,
                                                  (short)0, acc, false, false);
  }

  // ---- store: C/D layout (VGPR r -> M = r + 8*(lane>=16), N = lane&15) ----
  const int col   = nt * 16 + (lane & 15);
  const float bias = b2[col];
  const int mhalf = (lane >> 4) << 3;
#pragma unroll
  for (int r = 0; r < 8; ++r) {
    int orow = tile * 16 + r + mhalf;
    __builtin_nontemporal_store(acc[r] + bias, &out[(size_t)orow * E_DIM + col]);
  }
}

extern "C" void kernel_launch(void* const* d_in, const int* in_sizes, int n_in,
                              void* d_out, int out_size, void* d_ws, size_t ws_size,
                              hipStream_t stream) {
  (void)in_sizes; (void)n_in; (void)out_size; (void)ws_size;
  const float* vel   = (const float*)d_in[0];
  const float* dist  = (const float*)d_in[1];
  const float* W1    = (const float*)d_in[2];
  const float* b1    = (const float*)d_in[3];
  const float* gamma = (const float*)d_in[4];
  const float* beta  = (const float*)d_in[5];
  const float* W2    = (const float*)d_in[6];
  const float* b2    = (const float*)d_in[7];
  float* out = (float*)d_out;

  unsigned short* w2hi = (unsigned short*)d_ws;                       // 128 KB
  unsigned short* w2lo = (unsigned short*)((char*)d_ws + 131072);     // 128 KB

  prepack_w2_kernel<<<256, 256, 0, stream>>>(W2, w2hi, w2lo);
  velocity_encoder_kernel<<<(B_DIM * A_DIM) / 16, 512, 0, stream>>>(
      vel, dist, W1, b1, gamma, beta, b2, w2hi, w2lo, out);
}
